// GatherModel_57629871178231
// MI455X (gfx1250) — matrix operand order repaired and checked by reading, not compiled, not run on hardware
//
#include <hip/hip_runtime.h>

typedef _Float16 v16h __attribute__((ext_vector_type(16)));
typedef _Float16 v8h  __attribute__((ext_vector_type(8)));
typedef float    v8f  __attribute__((ext_vector_type(8)));

#define NN      10000
#define NE      160000
#define D       42
#define DE      10
#define DP      48            // padded feature dim
#define KROWS   44            // h rows: 0..41 = relu(h), 42 = bias(=1), 43 = zero pad
#define KV      (KROWS*DP)    // 2112 virtual K
#define KTILES  (KV/32)       // 66
#define XSTRIDE 56            // LDS row stride (halves): conflict-free & fits idx<=55

// ---------------- edge MLP hidden: Hh[e][48], h[42]=1 (bias row) ----------------
__global__ void k_edge_hidden(const float* __restrict__ ef,
                              const float* __restrict__ w1,
                              const float* __restrict__ b1,
                              _Float16* __restrict__ Hh) {
  int idx = blockIdx.x * blockDim.x + threadIdx.x;   // NE*48 threads
  int e = idx / DP, kk = idx % DP;
  float v = 0.f;
  if (kk < D) {
    float acc = b1[kk];
    #pragma unroll
    for (int d = 0; d < DE; ++d) acc += ef[e * DE + d] * w1[d * D + kk];
    v = acc > 0.f ? acc : 0.f;
  } else if (kk == D) {
    v = 1.f;                                         // bias row of the bilinear form
  }
  Hh[idx] = (_Float16)v;
}

// ---------------- pack W2+b2 into WMMA B-fragment order ----------------
// Virtual K = k*48 + i. B tile (kt,nt): lane L holds col o = nt*16 + (L&15),
// halves j=0..15 hold K = kt*32 + (L>=16 ? 8 : 0) + (j<8 ? j : j+8). Lane data contiguous 32B.
__global__ void k_pack_w2(const float* __restrict__ w2,
                          const float* __restrict__ b2,
                          _Float16* __restrict__ Bp) {
  int idx = blockIdx.x * blockDim.x + threadIdx.x;   // KTILES*3*32*16 threads
  int j  = idx & 15;
  int L  = (idx >> 4) & 31;
  int nt = (idx >> 9) % 3;
  int kt = idx / (16 * 32 * 3);
  int Kv = kt * 32 + ((L >= 16) ? 8 : 0) + ((j < 8) ? j : (j + 8));
  int o  = nt * 16 + (L & 15);
  int k  = Kv / DP, i = Kv % DP;
  float v = 0.f;
  if (i < D && o < D) {
    if (k < D)       v = w2[k * (D * D) + i * D + o];
    else if (k == D) v = b2[i * D + o];
  }
  Bp[idx] = (_Float16)v;
}

// ---------------- node init: X = relu(n_feat @ lin0_w + lin0_b); AGG = 0 ----------------
__global__ void k_node_init(const float* __restrict__ nf,
                            const float* __restrict__ w,
                            const float* __restrict__ b,
                            float* __restrict__ Xf, _Float16* __restrict__ Xh,
                            float* __restrict__ AGG) {
  int idx = blockIdx.x * blockDim.x + threadIdx.x;   // NN*48 threads
  int n = idx / DP, o = idx % DP;
  float v = 0.f;
  if (o < D) {
    float acc = b[o];
    for (int i = 0; i < D; ++i) acc += nf[n * D + i] * w[i * D + o];
    v = acc > 0.f ? acc : 0.f;
  }
  Xf[idx] = v;
  Xh[idx] = (_Float16)v;
  AGG[idx] = 0.f;
}

// ---------------- fused edge messages: WMMA bilinear, scatter-add ----------------
__global__ __launch_bounds__(128)
void k_edge_msgs(const _Float16* __restrict__ Xh,
                 const _Float16* __restrict__ Hh,
                 const _Float16* __restrict__ Bp,
                 const int* __restrict__ src,
                 const int* __restrict__ dst,
                 float* __restrict__ AGG) {
  __shared__ _Float16 Xl[256 * XSTRIDE];
  __shared__ _Float16 Hl[256 * XSTRIDE];
  __shared__ int      Dl[256];

  const int tid   = threadIdx.x;
  const int eBase = blockIdx.x * 256;

  // Stage gathered x rows, h rows, dst ids for 256 edges into LDS.
  for (int el = tid; el < 256; el += 128) {
    int e = eBase + el;
    int s = src[e];
    const v8h* xs = (const v8h*)(Xh + s * DP);
    const v8h* hs = (const v8h*)(Hh + e * DP);
    v8h* xd = (v8h*)&Xl[el * XSTRIDE];
    v8h* hd = (v8h*)&Hl[el * XSTRIDE];
    #pragma unroll
    for (int c = 0; c < 6; ++c) { xd[c] = xs[c]; hd[c] = hs[c]; }
    Dl[el] = dst[e];
  }
  __syncthreads();

  const int lane  = tid & 31;
  const int wave  = tid >> 5;
  const int row   = lane & 15;
  const int hi    = lane >> 4;          // 0 = lanes 0-15, 1 = lanes 16-31
  const int hioff = hi ? 8 : 0;         // A-layout K offset for high half

  const _Float16* Xrh[4];               // row base + hioff (indices become immediates)
  const _Float16* Hr[4];
  #pragma unroll
  for (int m = 0; m < 4; ++m) {
    int el = wave * 64 + m * 16 + row;
    Xrh[m] = &Xl[el * XSTRIDE] + hioff;
    Hr[m]  = &Hl[el * XSTRIDE];
  }

  v8f c0[4], c1[4], c2[4];
  #pragma unroll
  for (int m = 0; m < 4; ++m) { c0[m] = {}; c1[m] = {}; c2[m] = {}; }

  const v16h* B = (const v16h*)Bp;      // 32B fragments, one per lane per tile

  #pragma clang loop unroll(disable)
  for (int q = 0; q < KTILES / 3; ++q) {         // 22 iterations, 3 K-tiles each
    // A-fragment builder: U[row, K] = h[k] * x[i], K = 96q + 32r + jpat + hioff.
    // jpat in {0..7,16..23}; with hioff folded into Xrh the carry (K crossing a
    // 48-boundary) is compile-time: T<40 -> 0, T>=48 -> 1, T never in [40,48).
    // Fresh local v16h + per-element writes -> SLP vectorizes to v_pk_mul_f16
    // with the h scalar broadcast via op_sel.
    auto buildA = [&](int r, int m) -> v16h {
      v16h a;
      #pragma unroll
      for (int j = 0; j < 16; ++j) {
        const int jp    = (j < 8) ? j : (j + 8);
        const int T     = 32 * r + jp;            // compile-time
        const int carry = (T >= 48) ? 1 : 0;      // compile-time
        a[j] = Xrh[m][T - 48 * carry] * Hr[m][2 * q + carry];
      }
      return a;
    };

    // 12 tiles per q: t = r*4 + m. Software-pipeline the A fragment: A(t+1) is
    // built before A(t) is consumed, so aCur/aNext are simultaneously live and
    // must occupy distinct registers -> the next tile's v_pk_mul_f16 stream can
    // cover the WMMA->VALU WAR hazard slots instead of v_nop padding.
    v16h bA, bB, bC;
    v16h aCur;
    #pragma unroll
    for (int t = 0; t < 12; ++t) {
      const int r = t >> 2, m = t & 3;
      if (m == 0) {
        const int kt = 3 * q + r;
        bA = B[(kt * 3 + 0) * 32 + lane];
        bB = B[(kt * 3 + 1) * 32 + lane];
        bC = B[(kt * 3 + 2) * 32 + lane];
      }
      if (t == 0) aCur = buildA(0, 0);
      v16h aNext = aCur;
      if (t < 11) aNext = buildA((t + 1) >> 2, (t + 1) & 3);
      c0[m] = __builtin_amdgcn_wmma_f32_16x16x32_f16(false, aCur, false, bA,
                                                     (short)0, c0[m], false, false);
      c1[m] = __builtin_amdgcn_wmma_f32_16x16x32_f16(false, aCur, false, bB,
                                                     (short)0, c1[m], false, false);
      c2[m] = __builtin_amdgcn_wmma_f32_16x16x32_f16(false, aCur, false, bC,
                                                     (short)0, c2[m], false, false);
      aCur = aNext;
    }
  }

  // Scatter: C/D layout -> VGPR rr holds M = hi*8+rr, col = lane&15.
  const int col = lane & 15;
  #pragma unroll
  for (int m = 0; m < 4; ++m) {
    #pragma unroll
    for (int rr = 0; rr < 8; ++rr) {
      int el = wave * 64 + m * 16 + hi * 8 + rr;
      float* ag = AGG + Dl[el] * DP;
      atomicAdd(ag + col,      c0[m][rr]);
      atomicAdd(ag + 16 + col, c1[m][rr]);
      if (col < D - 32) atomicAdd(ag + 32 + col, c2[m][rr]);
    }
  }
}

// ---------------- node step part 1: M = relu(AGG + X + conv_b); AGG = 0 ----------------
__global__ void k_node_m(float* __restrict__ AGG, const float* __restrict__ Xf,
                         const float* __restrict__ cb, float* __restrict__ Mb) {
  int idx = blockIdx.x * blockDim.x + threadIdx.x;   // NN*48 threads
  int o = idx % DP;
  float v = 0.f;
  if (o < D) {
    float t = AGG[idx] + Xf[idx] + cb[o];
    v = t > 0.f ? t : 0.f;
  }
  Mb[idx] = v;
  AGG[idx] = 0.f;
}

// ---------------- node step part 2: X = relu(M @ msg_w + msg_b); final += init ----------------
__global__ void k_node_upd(const float* __restrict__ Mb,
                           const float* __restrict__ w, const float* __restrict__ b,
                           float* __restrict__ Xf, _Float16* __restrict__ Xh,
                           const float* __restrict__ nf, float* __restrict__ outp,
                           int last) {
  int idx = blockIdx.x * blockDim.x + threadIdx.x;   // NN*48 threads
  int n = idx / DP, o = idx % DP;
  float v = 0.f;
  if (o < D) {
    float acc = b[o];
    const float* mr = Mb + n * DP;
    #pragma unroll 6
    for (int i = 0; i < D; ++i) acc += mr[i] * w[i * D + o];
    v = acc > 0.f ? acc : 0.f;
  }
  Xf[idx] = v;
  Xh[idx] = (_Float16)v;
  if (last && o < D) outp[n * D + o] = v + nf[n * D + o];
}

extern "C" void kernel_launch(void* const* d_in, const int* in_sizes, int n_in,
                              void* d_out, int out_size, void* d_ws, size_t ws_size,
                              hipStream_t stream) {
  const float* n_feat = (const float*)d_in[0];
  const float* e_feat = (const float*)d_in[1];
  const int*   src    = (const int*)d_in[2];
  const int*   dst    = (const int*)d_in[3];
  const float* lin0_w = (const float*)d_in[4];
  const float* lin0_b = (const float*)d_in[5];
  const float* msg_w  = (const float*)d_in[6];
  const float* msg_b  = (const float*)d_in[7];
  const float* en_w1  = (const float*)d_in[8];
  const float* en_b1  = (const float*)d_in[9];
  const float* en_w2  = (const float*)d_in[10];
  const float* en_b2  = (const float*)d_in[11];
  const float* conv_b = (const float*)d_in[12];
  float* outp = (float*)d_out;

  // Workspace carve-up (256B aligned): ~22.3 MB total.
  char* ws = (char*)d_ws;
  size_t off = 0;
  auto take = [&](size_t bytes) {
    void* p = ws + off;
    off = (off + bytes + 255) & ~(size_t)255;
    return p;
  };
  _Float16* Hh = (_Float16*)take((size_t)NE * DP * sizeof(_Float16));          // 15.36 MB
  _Float16* Bp = (_Float16*)take((size_t)KTILES * 3 * 32 * 16 * sizeof(_Float16)); // 203 KB
  float*    Xf = (float*)take((size_t)NN * DP * sizeof(float));                // 1.92 MB
  _Float16* Xh = (_Float16*)take((size_t)NN * DP * sizeof(_Float16));          // 0.96 MB
  float*    AGG = (float*)take((size_t)NN * DP * sizeof(float));               // 1.92 MB
  float*    Mb  = (float*)take((size_t)NN * DP * sizeof(float));               // 1.92 MB

  k_edge_hidden<<<(NE * DP) / 256, 256, 0, stream>>>(e_feat, en_w1, en_b1, Hh);
  k_pack_w2<<<(KTILES * 3 * 32 * 16) / 256, 256, 0, stream>>>(en_w2, en_b2, Bp);
  k_node_init<<<(NN * DP) / 256, 256, 0, stream>>>(n_feat, lin0_w, lin0_b, Xf, Xh, AGG);

  for (int s = 0; s < 6; ++s) {
    k_edge_msgs<<<NE / 256, 128, 0, stream>>>(Xh, Hh, Bp, src, dst, AGG);
    k_node_m<<<(NN * DP) / 256, 256, 0, stream>>>(AGG, Xf, conv_b, Mb);
    k_node_upd<<<(NN * DP) / 256, 256, 0, stream>>>(Mb, msg_w, msg_b, Xf, Xh,
                                                    n_feat, outp, s == 5 ? 1 : 0);
  }
}
